// UCCModel_10728828305702
// MI455X (gfx1250) — matrix operand order repaired and checked by reading, not compiled, not run on hardware
//
#include <hip/hip_runtime.h>
#include <hip/hip_bf16.h>
#include <math.h>

// ---------------------------------------------------------------------------
// Fragment layouts for V_WMMA_F32_16X16X32_F16 (wave32), as used in this file:
//   A: 16x32 f16; lane l holds row (l&15);
//      elems j=0..15 -> K = (l>>4)*16 + j (contiguous 32B)
//   B: 32x16 f16; lane l holds column (l&15); K = (l>>4)*16 + j (contig 32B)
//   C/D: 16x16 f32; elem r of lane l -> M = (l>>4)*8 + r, N = l&15
// ---------------------------------------------------------------------------
typedef __attribute__((ext_vector_type(16))) _Float16 v16h;
typedef __attribute__((ext_vector_type(8)))  _Float16 v8h;
typedef __attribute__((ext_vector_type(8)))  float    v8f;
typedef __attribute__((ext_vector_type(4)))  unsigned int v4u;
typedef __attribute__((ext_vector_type(8)))  int      v8i;
typedef __attribute__((ext_vector_type(4)))  int      v4i;

#if defined(__HIP_DEVICE_COMPILE__) && __has_builtin(__builtin_amdgcn_tensor_load_to_lds) && __has_builtin(__builtin_amdgcn_s_wait_tensorcnt)
#define HAVE_TDM 1
#else
#define HAVE_TDM 0
#endif

static __device__ __forceinline__ v16h join8(v8h lo, v8h hi) {
  v16h r;
#pragma unroll
  for (int j = 0; j < 8; ++j) { r[j] = lo[j]; r[j + 8] = hi[j]; }
  return r;
}

#if HAVE_TDM
// 1-D contiguous TDM copy: global f16[elems] -> LDS. elems <= 65535.
// D# group0: count=1 | lds_addr | global_addr | type=2 ("image")
// D# group1: data_size=1(2B), tensor_dim0=elems, tile_dim0=elems, tile_dim1=1
static __device__ __forceinline__ void tdm_load_1d(void* lds, const void* gsrc,
                                                   unsigned elems) {
  unsigned long long ga = (unsigned long long)gsrc;
  v4u g0;
  g0[0] = 1u;                                            // count=1, user mode
  g0[1] = (unsigned)(size_t)lds;                         // LDS byte offset (low 32b of generic ptr)
  g0[2] = (unsigned)ga;                                  // global_addr[31:0]
  g0[3] = (unsigned)((ga >> 32) & 0x01ffffffu) | (2u << 30);  // addr[56:32] | type=2
  v8i g1;
  g1[0] = (int)(1u << 16);                               // data_size=1 (2 bytes)
  g1[1] = (int)((elems & 0xffffu) << 16);                // tensor_dim0[15:0]
  g1[2] = (int)((elems >> 16) & 0xffffu);                // tensor_dim0[31:16]
  g1[3] = (int)((elems & 0xffffu) << 16);                // tile_dim0 = elems
  g1[4] = 1;                                             // tile_dim1 = 1
  g1[5] = (int)elems;                                    // tensor_dim0_stride lo
  g1[6] = 0;
  g1[7] = 0;
  v4i gz = {};
#if __clang_major__ >= 23
  v8i gz8 = {};
  __builtin_amdgcn_tensor_load_to_lds(g0, g1, gz, gz, gz8, 0);
#else
  __builtin_amdgcn_tensor_load_to_lds(g0, g1, gz, gz, 0);
#endif
}
#endif

// im2col gather of one B element (handles transposed conv via lhs-dilation)
template <int KS, int LDIL>
static __device__ __forceinline__ _Float16 im2col_val(
    const _Float16* __restrict__ inp, int k, int Kreal, int oy, int ox,
    int stride, int pad, int Hin, int Win) {
  _Float16 v = (_Float16)0.f;
  if (k < Kreal) {
    int c  = k / (KS * KS);
    int r2 = k - c * (KS * KS);
    int ky = r2 / KS;
    int kx = r2 - ky * KS;
    int ty = oy * stride + ky - pad;
    int tx = ox * stride + kx - pad;
    if (ty >= 0 && tx >= 0 && (ty % LDIL) == 0 && (tx % LDIL) == 0) {
      int iy = ty / LDIL, ix = tx / LDIL;
      if (iy < Hin && ix < Win) v = inp[((size_t)c * Hin + iy) * Win + ix];
    }
  }
  return v;
}

// ---------------------------------------------------------------------------
// Implicit-GEMM convolution.
// STAGED=false: 1 wave per 16(Cout)x16(pixel) tile, direct global gather
//               (used when Mtiles==1: no reuse to exploit).
// STAGED=true : block = 2 waves = 2 M-tiles sharing one (img, pixel-tile).
//               A panel staged to LDS via TDM (or coop copy fallback),
//               B panel im2col-gathered once into LDS in fragment layout
//               [kstep][col][32], then ds_load_b128 fragment reads.
// EXEC is all-ones at every WMMA (whole-wave control flow only).
// ---------------------------------------------------------------------------
template <int KS, int LDIL, bool STAGED>
__global__ void conv_wmma_kernel(const _Float16* __restrict__ A,
                                 const float* __restrict__ bias,
                                 const _Float16* __restrict__ in,
                                 _Float16* __restrict__ out16,
                                 float* __restrict__ out32,
                                 int Cin, int Hin, int Win,
                                 int Cout, int Hout, int Wout,
                                 int stride, int pad,
                                 int Kreal, int Kpad, int MtilesOrMtg,
                                 int relu, long total_waves) {
  const int lane = threadIdx.x & 31;
  const int col  = lane & 15;
  const int hsel = lane >> 4;
  const int HW = Hout * Wout;
  const int ntiles = HW >> 4;

  if constexpr (STAGED) {
    // ---- block decode: one block per (img, mt-group, pixel-tile) ----
    const int mtgCount = MtilesOrMtg;
    int grp = blockIdx.x;
    const int img = grp / (mtgCount * ntiles);
    int r = grp - img * mtgCount * ntiles;
    const int mtg = r / ntiles;
    const int nt  = r - mtg * ntiles;
    const int wave = threadIdx.x >> 5;            // 2 waves: wave == tile-in-group
    const int tid  = threadIdx.x;                 // blockDim == 64

    extern __shared__ char smem[];
    _Float16* sA = (_Float16*)smem;               // [32][Kpad]
    _Float16* sB = sA + 32 * Kpad;                // [Kpad/32][16][32]
    const unsigned aElems = 32u * (unsigned)Kpad;
    const _Float16* __restrict__ Apanel = A + (size_t)mtg * 32 * Kpad;

#if HAVE_TDM
    if (wave == 0) tdm_load_1d(sA, Apanel, aElems);     // async DMA, overlap w/ gather
#else
    for (int e = tid * 8; e < (int)aElems; e += 64 * 8)
      *(v8h*)(sA + e) = *(const v8h*)(Apanel + e);
#endif

    // ---- cooperative im2col of the B panel into LDS ----
    const _Float16* __restrict__ inp = in + (size_t)img * Cin * Hin * Win;
    const int pbase = nt * 16;
    for (int e = tid; e < Kpad * 16; e += 64) {
      int kstep = e >> 9;                 // /512
      int rr = e & 511;
      int n = rr >> 5, kw = rr & 31;
      int k = (kstep << 5) + kw;
      int p = pbase + n;
      int oy = p / Wout, ox = p - oy * Wout;
      sB[e] = im2col_val<KS, LDIL>(inp, k, Kreal, oy, ox, stride, pad, Hin, Win);
    }
#if HAVE_TDM
    if (wave == 0) __builtin_amdgcn_s_wait_tensorcnt(0);
#endif
    __syncthreads();

    // ---- compute: wave w owns M-tile (mtg*2 + w) ----
    const _Float16* sArow = sA + (size_t)(wave * 16 + col) * Kpad;
    v8f acc = {};
    for (int k0 = 0; k0 < Kpad; k0 += 32) {
      v16h af = join8(*(const v8h*)(sArow + k0 + hsel * 16),
                      *(const v8h*)(sArow + k0 + hsel * 16 + 8));
      const _Float16* bsrc = sB + (k0 >> 5) * 512 + col * 32 + hsel * 16;
      v16h bf = join8(*(const v8h*)bsrc, *(const v8h*)(bsrc + 8));
      acc = __builtin_amdgcn_wmma_f32_16x16x32_f16(false, af, false, bf,
                                                   (short)0, acc, false, false);
    }
    const int mt = mtg * 2 + wave;
    const int p  = pbase + col;
#pragma unroll
    for (int rr = 0; rr < 8; ++rr) {
      int m = mt * 16 + hsel * 8 + rr;
      if (m < Cout) {
        float v = acc[rr] + bias[m];
        if (relu) v = v > 0.f ? v : 0.f;
        size_t o = ((size_t)img * Cout + m) * HW + p;
        if (out16) out16[o] = (_Float16)v;
        if (out32) out32[o] = v;
      }
    }
  } else {
    // ---- direct path (Mtiles == 1) ----
    const int Mtiles = MtilesOrMtg;
    long gid = (long)blockIdx.x * (blockDim.x >> 5) + (threadIdx.x >> 5);
    if (gid >= total_waves) return;               // uniform per wave
    const int img = (int)(gid / ((long)Mtiles * ntiles));
    int rem = (int)(gid - (long)img * Mtiles * ntiles);
    const int mt = rem / ntiles;
    const int nt = rem - mt * ntiles;
    const int p  = nt * 16 + col;
    const int oy = p / Wout;
    const int ox = p - oy * Wout;
    const _Float16* __restrict__ inp  = in + (size_t)img * Cin * Hin * Win;
    const _Float16* __restrict__ Arow = A + (size_t)(mt * 16 + col) * Kpad;

    v8f acc = {};
    for (int k0 = 0; k0 < Kpad; k0 += 32) {
      v16h af = join8(*(const v8h*)(Arow + k0 + hsel * 16),
                      *(const v8h*)(Arow + k0 + hsel * 16 + 8));
      v16h bf;
      const int kb = k0 + hsel * 16;
#pragma unroll
      for (int j = 0; j < 16; ++j)
        bf[j] = im2col_val<KS, LDIL>(inp, kb + j, Kreal, oy, ox, stride, pad, Hin, Win);
      acc = __builtin_amdgcn_wmma_f32_16x16x32_f16(false, af, false, bf,
                                                   (short)0, acc, false, false);
    }
#pragma unroll
    for (int rr = 0; rr < 8; ++rr) {
      int m = mt * 16 + hsel * 8 + rr;
      if (m < Cout) {
        float v = acc[rr] + bias[m];
        if (relu) v = v > 0.f ? v : 0.f;
        size_t o = ((size_t)img * Cout + m) * HW + p;
        if (out16) out16[o] = (_Float16)v;
        if (out32) out32[o] = v;
      }
    }
  }
}

// ---------------------------------------------------------------------------
// Dense GEMM: D[n][m] = act(A[m][:] . B[n][:] + bias[m]).
// MT = M-tiles computed per wave, sharing each B fragment (B traffic / MT).
// A: f16 [Mtiles*16][Kpad]; B: f16 [N][ldb] rows zero-padded to Kpad.
// act: 0=none 1=relu 2=sigmoid. Outputs row-major [n][ldo].
// ---------------------------------------------------------------------------
template <int MT>
__global__ void gemm_wmma_kernel(const _Float16* __restrict__ A,
                                 const float* __restrict__ bias,
                                 const _Float16* __restrict__ B,
                                 _Float16* __restrict__ out16,
                                 float* __restrict__ out32,
                                 int Mreal, int ntN, int Kpad, int ldb,
                                 int ldo, int act, long total_waves) {
  long gid = (long)blockIdx.x * (blockDim.x >> 5) + (threadIdx.x >> 5);
  if (gid >= total_waves) return;
  const int lane = threadIdx.x & 31;
  const int col  = lane & 15;
  const int hsel = lane >> 4;
  const int mtb = (int)(gid / ntN) * MT;
  const int nt  = (int)(gid - (long)(gid / ntN) * ntN);

  const _Float16* __restrict__ Abase = A + (size_t)(mtb * 16 + col) * Kpad;
  const _Float16* __restrict__ Brow  = B + (size_t)(nt * 16 + col) * ldb;

  v8f acc[MT];
#pragma unroll
  for (int i = 0; i < MT; ++i) acc[i] = (v8f){};

  for (int k0 = 0; k0 < Kpad; k0 += 32) {
    v16h bf = join8(*(const v8h*)(Brow + k0 + hsel * 16),
                    *(const v8h*)(Brow + k0 + hsel * 16 + 8));
#pragma unroll
    for (int i = 0; i < MT; ++i) {
      const _Float16* Arow = Abase + (size_t)i * 16 * Kpad;
      v16h af = join8(*(const v8h*)(Arow + k0 + hsel * 16),
                      *(const v8h*)(Arow + k0 + hsel * 16 + 8));
      acc[i] = __builtin_amdgcn_wmma_f32_16x16x32_f16(false, af, false, bf,
                                                      (short)0, acc[i], false, false);
    }
  }
  const int n = nt * 16 + col;
#pragma unroll
  for (int i = 0; i < MT; ++i) {
#pragma unroll
    for (int r = 0; r < 8; ++r) {
      int m = (mtb + i) * 16 + hsel * 8 + r;
      if (m < Mreal) {
        float v = acc[i][r] + bias[m];
        if (act == 1) v = v > 0.f ? v : 0.f;
        else if (act == 2) v = 1.f / (1.f + __expf(-v));
        if (out16) out16[(size_t)n * ldo + m] = (_Float16)v;
        if (out32) out32[(size_t)n * ldo + m] = v;
      }
    }
  }
}

// ---------------------------------------------------------------------------
// KDE: emb (Bk,N,F) f32 -> fd (Bk,ldo) f16; 11 bins 0..1, sigma=0.1 -> -50;
// Gaussian prefactor cancels under normalization. K padding zeroed.
// ---------------------------------------------------------------------------
__global__ void kde_kernel(const float* __restrict__ emb,
                           _Float16* __restrict__ fd,
                           int Bk, int N, int F, int ldo) {
  int idx = blockIdx.x * blockDim.x + threadIdx.x;
  if (idx >= Bk * F) return;
  int b = idx / F, f = idx - b * F;
  float s[11];
#pragma unroll
  for (int i = 0; i < 11; ++i) s[i] = 0.f;
  for (int n = 0; n < N; ++n) {
    float v = emb[((size_t)b * N + n) * F + f];
#pragma unroll
    for (int i = 0; i < 11; ++i) {
      float d = 0.1f * (float)i - v;
      s[i] += __expf(-50.f * d * d);
    }
  }
  float tot = 0.f;
#pragma unroll
  for (int i = 0; i < 11; ++i) tot += s[i];
  float inv = 1.f / tot;
#pragma unroll
  for (int i = 0; i < 11; ++i)
    fd[(size_t)b * ldo + f * 11 + i] = (_Float16)(s[i] * inv);
  if (f < ldo - 11 * F) fd[(size_t)b * ldo + 11 * F + f] = (_Float16)0.f;
}

// ---------------------------------------------------------------------------
// Weight packing / casting (tiny, once per launch)
// ---------------------------------------------------------------------------
__global__ void pack_conv_w_kernel(const float* __restrict__ w, _Float16* __restrict__ A,
                                   int Cout, int Cin, int kk, int Kpad, long total) {
  long i = (long)blockIdx.x * blockDim.x + threadIdx.x;
  if (i >= total) return;
  int m = (int)(i / Kpad), k = (int)(i - (long)m * Kpad);
  float v = 0.f;
  int K = Cin * kk;
  if (m < Cout && k < K) {
    int c = k / kk, r = k - c * kk;
    v = w[((size_t)m * Cin + c) * kk + r];
  }
  A[i] = (_Float16)v;
}

__global__ void pack_convT_w_kernel(const float* __restrict__ w, _Float16* __restrict__ A,
                                    int Cin, int Cout, int Kpad, long total) {
  long i = (long)blockIdx.x * blockDim.x + threadIdx.x;
  if (i >= total) return;
  int m = (int)(i / Kpad), k = (int)(i - (long)m * Kpad);
  float v = 0.f;
  int K = Cin * 9;
  if (m < Cout && k < K) {
    int c = k / 9, r = k - c * 9;
    v = w[((size_t)c * Cout + m) * 9 + 8 - r];   // spatially flipped tap
  }
  A[i] = (_Float16)v;
}

__global__ void pack_dense_w_kernel(const float* __restrict__ w, _Float16* __restrict__ A,
                                    int M, int K, int Kpad, long total) {
  long i = (long)blockIdx.x * blockDim.x + threadIdx.x;
  if (i >= total) return;
  int m = (int)(i / Kpad), k = (int)(i - (long)m * Kpad);
  A[i] = (_Float16)((m < M && k < K) ? w[(size_t)m * K + k] : 0.f);
}

__global__ void cast_f16_kernel(const float* __restrict__ src, _Float16* __restrict__ dst,
                                long n) {
  long i = (long)blockIdx.x * blockDim.x + threadIdx.x;
  if (i < n) dst[i] = (_Float16)src[i];
}

// ---------------------------------------------------------------------------
extern "C" void kernel_launch(void* const* d_in, const int* in_sizes, int n_in,
                              void* d_out, int out_size, void* d_ws, size_t ws_size,
                              hipStream_t stream) {
  (void)in_sizes; (void)n_in; (void)out_size; (void)ws_size;
  const float* x   = (const float*)d_in[0];
  const float* ew1 = (const float*)d_in[1];  const float* eb1 = (const float*)d_in[2];
  const float* ew2 = (const float*)d_in[3];  const float* eb2 = (const float*)d_in[4];
  const float* ew3 = (const float*)d_in[5];  const float* eb3 = (const float*)d_in[6];
  const float* kw  = (const float*)d_in[7];  const float* kb  = (const float*)d_in[8];
  const float* dw1 = (const float*)d_in[9];  const float* db1 = (const float*)d_in[10];
  const float* dw2 = (const float*)d_in[11]; const float* db2 = (const float*)d_in[12];
  const float* dw3 = (const float*)d_in[13]; const float* db3 = (const float*)d_in[14];
  const float* dw4 = (const float*)d_in[15]; const float* db4 = (const float*)d_in[16];
  const float* f1w = (const float*)d_in[17]; const float* f1b = (const float*)d_in[18];
  const float* f2w = (const float*)d_in[19]; const float* f2b = (const float*)d_in[20];
  const float* f3w = (const float*)d_in[21]; const float* f3b = (const float*)d_in[22];
  float* outf = (float*)d_out;               // [0,256): logits, [256,...): decoded NCHW

  // ---- workspace carve-up (~222 MB peak, 256B aligned) ----
  char* ws = (char*)d_ws;
  size_t off = 0;
  auto alloc = [&](size_t bytes) -> void* {
    off = (off + 255) & ~(size_t)255;
    void* p = ws + off;
    off += bytes;
    return p;
  };
  _Float16* Aw1 = (_Float16*)alloc((size_t)16 * 32 * 2);
  _Float16* Aw2 = (_Float16*)alloc((size_t)32 * 160 * 2);
  _Float16* Aw3 = (_Float16*)alloc((size_t)64 * 288 * 2);
  _Float16* Ad1 = (_Float16*)alloc((size_t)64 * 576 * 2);
  _Float16* Ad2 = (_Float16*)alloc((size_t)32 * 576 * 2);
  _Float16* Ad3 = (_Float16*)alloc((size_t)16 * 288 * 2);
  _Float16* Ad4 = (_Float16*)alloc((size_t)16 * 32 * 2);
  _Float16* Akw = (_Float16*)alloc((size_t)112 * 4096 * 2);
  _Float16* Af1 = (_Float16*)alloc((size_t)512 * 1216 * 2);
  _Float16* Af2 = (_Float16*)alloc((size_t)256 * 512 * 2);
  _Float16* Af3 = (_Float16*)alloc((size_t)16 * 256 * 2);
  // Activation ping-pong: R1 <-> R2; h persists for the embedding GEMM.
  _Float16* R1 = (_Float16*)alloc((size_t)2048 * 16 * 1024 * 2);  // e1/d1/d3
  _Float16* R2 = (_Float16*)alloc((size_t)2048 * 32 * 1024 * 2);  // a0/e2/d2
  _Float16* h  = (_Float16*)alloc((size_t)2048 * 64 * 64 * 2);
  float*    emb = (float*)alloc((size_t)2048 * 110 * 4);
  _Float16* fd  = (_Float16*)alloc((size_t)64 * 1216 * 2);
  _Float16* m1  = (_Float16*)alloc((size_t)64 * 512 * 2);
  _Float16* m2  = (_Float16*)alloc((size_t)64 * 256 * 2);

  auto cdiv = [](long a, long b) { return (unsigned)((a + b - 1) / b); };
  const int BN = 2048;
  long t;

  // ---- pack weights to f16 GEMM layouts, cast input ----
  t = 16 * 32;        pack_conv_w_kernel <<<cdiv(t,256),256,0,stream>>>(ew1, Aw1, 16, 3, 9, 32, t);
  t = 32 * 160;       pack_conv_w_kernel <<<cdiv(t,256),256,0,stream>>>(ew2, Aw2, 32, 16, 9, 160, t);
  t = 64 * 288;       pack_conv_w_kernel <<<cdiv(t,256),256,0,stream>>>(ew3, Aw3, 64, 32, 9, 288, t);
  t = 64 * 576;       pack_convT_w_kernel<<<cdiv(t,256),256,0,stream>>>(dw1, Ad1, 64, 64, 576, t);
  t = 32 * 576;       pack_convT_w_kernel<<<cdiv(t,256),256,0,stream>>>(dw2, Ad2, 64, 32, 576, t);
  t = 16 * 288;       pack_convT_w_kernel<<<cdiv(t,256),256,0,stream>>>(dw3, Ad3, 32, 16, 288, t);
  t = 16 * 32;        pack_conv_w_kernel <<<cdiv(t,256),256,0,stream>>>(dw4, Ad4, 3, 16, 1, 32, t);
  t = 112L * 4096;    pack_dense_w_kernel<<<cdiv(t,256),256,0,stream>>>(kw,  Akw, 110, 4096, 4096, t);
  t = 512L * 1216;    pack_dense_w_kernel<<<cdiv(t,256),256,0,stream>>>(f1w, Af1, 512, 1210, 1216, t);
  t = 256L * 512;     pack_dense_w_kernel<<<cdiv(t,256),256,0,stream>>>(f2w, Af2, 256, 512, 512, t);
  t = 16L * 256;      pack_dense_w_kernel<<<cdiv(t,256),256,0,stream>>>(f3w, Af3, 4, 256, 256, t);
  t = (long)BN * 3 * 1024; cast_f16_kernel<<<cdiv(t,256),256,0,stream>>>(x, R2, t);   // a0

  // ---- encoder ----
  t = (long)BN * 64;   // conv1: Mtiles=1, direct path
  conv_wmma_kernel<3,1,false><<<cdiv(t,8),256,0,stream>>>(Aw1, eb1, R2, R1, nullptr,
      3, 32, 32, 16, 32, 32, 1, 1, 27, 32, 1, 1, t);
  // conv2: Mtiles=2 -> 1 mt-group; block=64 thr; smem = 96*Kpad bytes
  conv_wmma_kernel<3,1,true><<<(unsigned)(BN * 1 * 16), 64, 96 * 160, stream>>>(
      Aw2, eb2, R1, R2, nullptr, 16, 32, 32, 32, 16, 16, 2, 1, 144, 160, 1, 1, 0);
  // conv3: Mtiles=4 -> 2 mt-groups
  conv_wmma_kernel<3,1,true><<<(unsigned)(BN * 2 * 4), 64, 96 * 288, stream>>>(
      Aw3, eb3, R2, h, nullptr, 32, 16, 16, 64, 8, 8, 2, 1, 288, 288, 2, 1, 0);

  // ---- decoder (transposed convs as lhs-dilated convs) ----
  conv_wmma_kernel<3,2,true><<<(unsigned)(BN * 2 * 16), 64, 96 * 576, stream>>>(
      Ad1, db1, h, R1, nullptr, 64, 8, 8, 64, 16, 16, 1, 1, 576, 576, 2, 1, 0);
  conv_wmma_kernel<3,2,true><<<(unsigned)(BN * 1 * 64), 64, 96 * 576, stream>>>(
      Ad2, db2, R1, R2, nullptr, 64, 16, 16, 32, 32, 32, 1, 1, 576, 576, 1, 1, 0);
  t = (long)BN * 64;   // convT3: Mtiles=1, direct
  conv_wmma_kernel<3,1,false><<<cdiv(t,8),256,0,stream>>>(Ad3, db3, R2, R1, nullptr,
      32, 32, 32, 16, 32, 32, 1, 1, 288, 288, 1, 1, t);
  t = (long)BN * 64;   // conv4 1x1 -> f32 decoded image into d_out+256
  conv_wmma_kernel<1,1,false><<<cdiv(t,8),256,0,stream>>>(Ad4, db4, R1, nullptr, outf + 256,
      16, 32, 32, 3, 32, 32, 1, 0, 16, 32, 1, 0, t);

  // ---- embedding GEMM + sigmoid: all 7 M-tiles per wave (B traffic /7) ----
  t = 128;
  gemm_wmma_kernel<7><<<cdiv(t,8),256,0,stream>>>(Akw, kb, h, nullptr, emb,
      110, 128, 4096, 4096, 110, 2, t);

  // ---- KDE -> fd (64,1216) f16, zero padded ----
  t = 64L * 110;
  kde_kernel<<<cdiv(t,256),256,0,stream>>>(emb, fd, 64, 32, 110, 1216);

  // ---- MLP classifier ----
  t = 32L * 4;
  gemm_wmma_kernel<1><<<cdiv(t,8),256,0,stream>>>(Af1, f1b, fd, m1, nullptr,
      512, 4, 1216, 1216, 512, 1, t);
  t = 16L * 4;
  gemm_wmma_kernel<1><<<cdiv(t,8),256,0,stream>>>(Af2, f2b, m1, m2, nullptr,
      256, 4, 512, 512, 256, 1, t);
  t = 1L * 4;
  gemm_wmma_kernel<1><<<cdiv(t,8),256,0,stream>>>(Af3, f3b, m2, nullptr, outf,
      4, 4, 256, 256, 4, 0, t);
}
